// ForwardBPTTCell_68066641707155
// MI455X (gfx1250) — matrix operand order repaired and checked by reading, not compiled, not run on hardware
//
#include <hip/hip_runtime.h>
#include <hip/hip_bf16.h>

// ---------------------------------------------------------------------------
// ForwardBPTTCell for MI455X (gfx1250, wave32, WMMA, async LDS->global DMA).
// One workgroup per batch sequence (B=4), 256 threads (8 wave32 waves).
// - prod_jac @ jac (128x128x128 fp32) on V_WMMA_F32_16X16X4_F32.
// - Per-step 64KB Jacobian output streamed to HBM with
//   GLOBAL_STORE_ASYNC_FROM_LDS_B128, overlapped with the WMMA product and
//   the Frobenius-norm reduction; ASYNCcnt waited only before the in-place
//   Gaussian elimination destroys jac.
// ---------------------------------------------------------------------------

typedef __attribute__((ext_vector_type(2)))  float    v2f;
typedef __attribute__((ext_vector_type(8)))  float    v8f;
typedef __attribute__((ext_vector_type(16))) _Float16 v16h;

#define HDIM 128
#define IDIM 64
#define ODIM 64
#define TLEN 512
#define BATCH 4
#define LDJ  132          // padded LDS row stride: 132 floats = 528B (16B aligned rows)
#define NTHREADS 256
#define LN_EPS 1e-6f

#if defined(__has_builtin)
#  if __has_builtin(__builtin_amdgcn_wmma_f32_16x16x4_f32)
#    define HAVE_WMMA_F32F32 1
#  endif
#endif

__device__ __forceinline__ float sigmoidf_(float x) {
    return 1.0f / (1.0f + __expf(-x));
}

__device__ __forceinline__ float blockReduceSum(float v, float* red) {
    const int tid = threadIdx.x;
    red[tid] = v;
    __syncthreads();
#pragma unroll
    for (int s = NTHREADS / 2; s > 0; s >>= 1) {
        if (tid < s) red[tid] += red[tid + s];
        __syncthreads();
    }
    float r = red[0];
    __syncthreads();
    return r;
}

__global__ void __launch_bounds__(NTHREADS)
fbptt_gru_kernel(const float* __restrict__ x,  const float* __restrict__ y,
                 const float* __restrict__ msk,
                 const float* __restrict__ Wi, const float* __restrict__ Wh,
                 const float* __restrict__ bi, const float* __restrict__ bh,
                 const float* __restrict__ gamma, const float* __restrict__ beta,
                 const float* __restrict__ Wo, const float* __restrict__ bo,
                 float* __restrict__ out_o,  float* __restrict__ h_o,
                 float* __restrict__ delta_o, float* __restrict__ inst_o,
                 float* __restrict__ dout_o, float* __restrict__ nh_o,
                 float* __restrict__ nd_o,   float* __restrict__ npj_o,
                 float* __restrict__ jac_o)
{
    extern __shared__ __align__(16) float smem[];
    float* jac  = smem;                 // 128*132
    float* pj   = jac  + HDIM * LDJ;    // 128*132  (running prod_jac)
    float* gi   = pj   + HDIM * LDJ;    // 384
    float* gh   = gi   + 3 * HDIM;      // 384
    float* h    = gh   + 3 * HDIM;      // 128 (carry h)
    float* nh   = h    + HDIM;          // 128 (new h)
    float* rg   = nh   + HDIM;          // 128 gate r
    float* zg   = rg   + HDIM;          // 128 gate z
    float* ng   = zg   + HDIM;          // 128 gate n
    float* hng  = ng   + HDIM;          // 128 h-side n preact
    float* dl   = hng  + HDIM;          // 128 delta carry
    float* il   = dl   + HDIM;          // 128 inst_delta carry
    float* rhs  = il   + HDIM;          // 128 solve rhs / new delta
    float* uu   = rhs  + HDIM;          // 128 layernorm-backward work
    float* outv = uu   + HDIM;          // 64
    float* dov  = outv + ODIM;          // 64
    float* red  = dov  + ODIM;          // 256 reduction scratch

    const int tid = threadIdx.x;
    const int b   = blockIdx.x;

    const float* xb = x   + (size_t)b * TLEN * IDIM;
    const float* yb = y   + (size_t)b * TLEN * ODIM;
    const float* mb = msk + (size_t)b * TLEN;

    // ---- carry init: h = 0, delta = 0, inst_delta = 0, prod_jac = I -------
    for (int e = tid; e < HDIM * HDIM; e += NTHREADS) {
        int r = e >> 7, c = e & 127;
        pj[r * LDJ + c] = (r == c) ? 1.0f : 0.0f;
    }
    for (int i = tid; i < HDIM; i += NTHREADS) { h[i] = 0.f; dl[i] = 0.f; il[i] = 0.f; }
    __syncthreads();

    for (int t = 0; t < TLEN; ++t) {
        const float* xt = xb + (size_t)t * IDIM;
        const float* yt = yb + (size_t)t * ODIM;
        const float  mt = mb[t];
        const size_t st = (size_t)(b * TLEN + t);

        if (t + 1 < TLEN)
            __builtin_prefetch(xb + (size_t)(t + 1) * IDIM, 0, 0); // global_prefetch_b8

        // ---- GRU pre-activations: gi = x@Wi + bi ; gh = h@Wh + bh ---------
        for (int o = tid; o < 3 * HDIM; o += NTHREADS) {
            float si = bi[o];
            for (int k = 0; k < IDIM; ++k) si += xt[k] * Wi[k * (3 * HDIM) + o];
            float sh = bh[o];
            for (int k = 0; k < HDIM; ++k) sh += h[k] * Wh[k * (3 * HDIM) + o];
            gi[o] = si; gh[o] = sh;
        }
        __syncthreads();

        // ---- gates + new h ------------------------------------------------
        for (int i = tid; i < HDIM; i += NTHREADS) {
            float rr = sigmoidf_(gi[i] + gh[i]);
            float zz = sigmoidf_(gi[HDIM + i] + gh[HDIM + i]);
            float hn = gh[2 * HDIM + i];
            float nn = tanhf(gi[2 * HDIM + i] + rr * hn);
            rg[i] = rr; zg[i] = zz; hng[i] = hn; ng[i] = nn;
            nh[i] = (1.f - zz) * nn + zz * h[i];
        }
        __syncthreads();

        // ---- analytic Jacobian:  jac[j][i] = d newh_i / d h_j (LDS only) --
        for (int e = tid; e < HDIM * HDIM; e += NTHREADS) {
            int j = e >> 7, i = e & 127;
            float ri = rg[i], zi = zg[i], ni = ng[i], hni = hng[i];
            const float* wrow = Wh + (size_t)j * (3 * HDIM);
            float dr = ri * (1.f - ri) * wrow[i];              // d r_i / d h_j
            float dz = zi * (1.f - zi) * wrow[HDIM + i];       // d z_i / d h_j
            float dn = (1.f - ni * ni) * (ri * wrow[2 * HDIM + i] + hni * dr);
            float v  = (h[i] - ni) * dz + (1.f - zi) * dn + ((i == j) ? zi : 0.f);
            jac[j * LDJ + i] = v;
        }
        __syncthreads();   // jac visible in LDS to all waves (implies dscnt drain)

        // ---- fire-and-forget: jac LDS -> global via async B128 DMA --------
        // 128 rows x 32 16B-chunks = 4096 chunks; 16 per thread. These drain
        // in the background under the WMMA product + norm below (ASYNCcnt).
        {
            float* jout = jac_o + st * (size_t)(HDIM * HDIM);
            for (int c = tid; c < HDIM * 32; c += NTHREADS) {
                const int row  = c >> 5;
                const int col4 = (c & 31) << 2;                 // float index, 16B chunks
                unsigned laddr =
                    (unsigned)(size_t)(const void*)(jac + row * LDJ + col4);
                unsigned long long gaddr =
                    (unsigned long long)(jout + row * HDIM + col4);
                asm volatile("global_store_async_from_lds_b128 %0, %1, off"
                             :: "v"(gaddr), "v"(laddr) : "memory");
            }
        }

        // ---- prod_jac = prod_jac @ jac  (WMMA, 64 tiles, 8 waves) ---------
        {
            const int lane = tid & 31, half = lane >> 4, l16 = lane & 15;
            const int tm   = tid >> 5;               // tile row = wave id
            v8f acc[8];
#ifdef HAVE_WMMA_F32F32
            for (int tn = 0; tn < 8; ++tn) {
                v8f c = {0.f, 0.f, 0.f, 0.f, 0.f, 0.f, 0.f, 0.f};
                const int row = tm * 16 + l16;
                const int col = tn * 16 + l16;
                for (int k0 = 0; k0 < HDIM; k0 += 4) {
                    const int kk = k0 + 2 * half;     // A 16x4: lanes 0-15 K0/K1, 16-31 K2/K3
                    v2f a, bb;
                    a.x  = pj[row * LDJ + kk];
                    a.y  = pj[row * LDJ + kk + 1];
                    bb.x = jac[kk * LDJ + col];       // B 4x16: VGPR r -> K = 2*half + r
                    bb.y = jac[(kk + 1) * LDJ + col];
                    c = __builtin_amdgcn_wmma_f32_16x16x4_f32(
                            false, a, false, bb, (short)0, c, false, false);
                }
                acc[tn] = c;
            }
#else
            // fallback: f32->f16 convert, V_WMMA_F32_16X16X32_F16 (codegen-confirmed)
            for (int tn = 0; tn < 8; ++tn) {
                v8f c = {0.f, 0.f, 0.f, 0.f, 0.f, 0.f, 0.f, 0.f};
                const int row = tm * 16 + l16;
                const int col = tn * 16 + l16;
                for (int k0 = 0; k0 < HDIM; k0 += 32) {
                    v16h a{}, bm{};
                    for (int hh = 0; hh < 16; ++hh) {
                        int ka = k0 + hh + 16 * half;
                        a[hh]  = (_Float16)pj[row * LDJ + ka];
                        int kb = k0 + hh + 16 * half;
                        bm[hh] = (_Float16)jac[kb * LDJ + col];
                    }
                    c = __builtin_amdgcn_wmma_f32_16x16x32_f16(
                            false, a, false, bm, (short)0, c, false, false);
                }
                acc[tn] = c;
            }
#endif
            __syncthreads();  // all waves done reading old pj
            for (int tn = 0; tn < 8; ++tn)
                for (int vr = 0; vr < 8; ++vr)   // D: (M = vr + 8*half, N = l16)
                    pj[(tm * 16 + vr + 8 * half) * LDJ + tn * 16 + l16] = acc[tn][vr];
        }
        __syncthreads();

        // ---- ||prod_jac||_F (async jac store still draining) --------------
        {
            float ss = 0.f;
            for (int e = tid; e < HDIM * HDIM; e += NTHREADS) {
                float v = pj[(e >> 7) * LDJ + (e & 127)];
                ss += v * v;
            }
            float tot = blockReduceSum(ss, red);
            if (tid == 0) npj_o[st] = sqrtf(tot);
        }

        // ---- new_delta = solve(jac, delta - inst_delta) -------------------
        for (int i = tid; i < HDIM; i += NTHREADS) rhs[i] = dl[i] - il[i];
        // jac is about to be destroyed in place: every wave drains its own
        // async stores, then barrier so all waves' DMA is complete.
        asm volatile("s_wait_asynccnt 0x0" ::: "memory");
        __syncthreads();
        for (int p = 0; p < HDIM; ++p) {                 // forward elimination
            const float inv = 1.f / jac[p * LDJ + p];
            for (int row = p + 1 + tid; row < HDIM; row += NTHREADS) {
                float f = jac[row * LDJ + p] * inv;
                for (int c2 = p + 1; c2 < HDIM; ++c2)
                    jac[row * LDJ + c2] -= f * jac[p * LDJ + c2];
                rhs[row] -= f * rhs[p];
            }
            __syncthreads();
        }
        for (int p = HDIM - 1; p >= 0; --p) {            // back substitution
            if (tid == 0) rhs[p] /= jac[p * LDJ + p];
            __syncthreads();
            const float rp = rhs[p];
            for (int row = tid; row < p; row += NTHREADS)
                rhs[row] -= jac[row * LDJ + p] * rp;
            __syncthreads();
        }

        // ---- delta / h outputs + norms ------------------------------------
        {
            float ds = 0.f;
            for (int i = tid; i < HDIM; i += NTHREADS) {
                float v = rhs[i];
                delta_o[st * HDIM + i] = v;
                ds += v * v;
            }
            float dtot = blockReduceSum(ds, red);
            if (tid == 0) nd_o[st] = sqrtf(dtot);

            float hs = 0.f;
            for (int i = tid; i < HDIM; i += NTHREADS) {
                h_o[st * HDIM + i] = h[i];               // pre-step carry h
                float v = nh[i];
                hs += v * v;
            }
            float htot = blockReduceSum(hs, red);
            if (tid == 0) nh_o[st] = sqrtf(htot);
        }

        // ---- layernorm stats on new h -------------------------------------
        float sm = 0.f;
        for (int i = tid; i < HDIM; i += NTHREADS) sm += nh[i];
        const float mu = blockReduceSum(sm, red) * (1.f / HDIM);
        float sv = 0.f;
        for (int i = tid; i < HDIM; i += NTHREADS) { float d0 = nh[i] - mu; sv += d0 * d0; }
        const float var  = blockReduceSum(sv, red) * (1.f / HDIM);
        const float invs = rsqrtf(var + LN_EPS);

        // ---- readout + delta_output ---------------------------------------
        for (int o = tid; o < ODIM; o += NTHREADS) {
            float s = bo[o];
            for (int k = 0; k < HDIM; ++k) {
                float xh = (nh[k] - mu) * invs;
                s += (xh * gamma[k] + beta[k]) * Wo[k * ODIM + o];
            }
            float d0 = mt * (s - yt[o]);
            outv[o] = s; dov[o] = d0;
            out_o[st * ODIM + o]  = s;
            dout_o[st * ODIM + o] = d0;
        }
        __syncthreads();

        // ---- inst_delta = dL/d(new_h) through layernorm + Wo --------------
        for (int k = tid; k < HDIM; k += NTHREADS) {
            float s = 0.f;
            for (int o = 0; o < ODIM; ++o) s += Wo[k * ODIM + o] * dov[o];
            uu[k] = gamma[k] * s;                        // dL/d xhat
        }
        __syncthreads();
        float s1l = 0.f, s2l = 0.f;
        for (int k = tid; k < HDIM; k += NTHREADS) {
            float xh = (nh[k] - mu) * invs;
            s1l += uu[k]; s2l += uu[k] * xh;
        }
        const float s1 = blockReduceSum(s1l, red) * (1.f / HDIM);
        const float s2 = blockReduceSum(s2l, red) * (1.f / HDIM);

        // ---- commit outputs + update carries ------------------------------
        for (int k = tid; k < HDIM; k += NTHREADS) {
            float xh  = (nh[k] - mu) * invs;
            float gkn = invs * (uu[k] - s1 - xh * s2);
            inst_o[st * HDIM + k] = gkn;
            il[k] = gkn;
            dl[k] = rhs[k];
            h[k]  = nh[k];
        }
        __syncthreads();
    }
}

extern "C" void kernel_launch(void* const* d_in, const int* in_sizes, int n_in,
                              void* d_out, int out_size, void* d_ws, size_t ws_size,
                              hipStream_t stream) {
    (void)in_sizes; (void)n_in; (void)d_ws; (void)ws_size; (void)out_size;

    const float* x     = (const float*)d_in[0];
    const float* y     = (const float*)d_in[1];
    const float* m     = (const float*)d_in[2];
    const float* Wi    = (const float*)d_in[3];
    const float* Wh    = (const float*)d_in[4];
    const float* bi    = (const float*)d_in[5];
    const float* bh    = (const float*)d_in[6];
    const float* gamma = (const float*)d_in[7];
    const float* beta  = (const float*)d_in[8];
    const float* Wo    = (const float*)d_in[9];
    const float* bo    = (const float*)d_in[10];

    float* out = (float*)d_out;
    size_t off = 0;
    float* out_o   = out + off; off += (size_t)BATCH * TLEN * ODIM;      // out
    float* h_o     = out + off; off += (size_t)BATCH * TLEN * HDIM;      // h (pre-step)
    float* delta_o = out + off; off += (size_t)BATCH * TLEN * HDIM;      // new_delta
    float* inst_o  = out + off; off += (size_t)BATCH * TLEN * HDIM;      // new_inst_delta
    float* dout_o  = out + off; off += (size_t)BATCH * TLEN * ODIM;      // delta_output
    float* nh_o    = out + off; off += (size_t)BATCH * TLEN;             // ||new_h||
    float* nd_o    = out + off; off += (size_t)BATCH * TLEN;             // ||new_delta||
    float* npj_o   = out + off; off += (size_t)BATCH * TLEN;             // ||new_prod_jac||
    float* jac_o   = out + off;                                          // jac

    const size_t smemFloats =
        2 * (size_t)HDIM * LDJ   // jac, prod_jac (132-float padded rows)
        + 2 * 3 * HDIM           // gi, gh
        + 8 * HDIM               // h, nh, r, z, n, hn, delta, inst
        + 2 * HDIM               // rhs, uu
        + 2 * ODIM               // outv, dov
        + NTHREADS;              // reduction scratch
    const size_t smemBytes = smemFloats * sizeof(float);

    hipFuncSetAttribute((const void*)fbptt_gru_kernel,
                        hipFuncAttributeMaxDynamicSharedMemorySize,
                        (int)smemBytes);

    fbptt_gru_kernel<<<dim3(BATCH), dim3(NTHREADS), smemBytes, stream>>>(
        x, y, m, Wi, Wh, bi, bh, gamma, beta, Wo, bo,
        out_o, h_o, delta_o, inst_o, dout_o, nh_o, nd_o, npj_o, jac_o);
}
